// MambaGlobalContext_58342835749268
// MI455X (gfx1250) — compile-verified, matrix operand order checked by reading, not run on hardware
//
#include <hip/hip_runtime.h>
#include <hip/hip_bf16.h>

// ---------------- problem constants ----------------
#define Bn      4
#define Cc      96
#define Hh      64
#define Ww      64
#define Ll      (Hh*Ww)          // 4096
#define NT      (Bn*Ll)          // 16384 tokens
#define NLAYERS 2
#define DSTATE  16
#define DIN     192              // EXPAND*C
#define D2IN    384              // 2*DIN
#define DCONV   4
#define DTRANK  6
#define XPJ     (DTRANK + 2*DSTATE)   // 38
#define EPSv    1e-5f
#define NCH     32               // scan chunks
#define CH      (Ll/NCH)         // 128 steps per chunk

typedef __attribute__((ext_vector_type(16))) _Float16 v16h;
typedef __attribute__((ext_vector_type(8)))  _Float16 v8h;
typedef __attribute__((ext_vector_type(8)))  float    v8f;

__device__ inline v8h ld8(const _Float16* p) {
    return *reinterpret_cast<const v8h*>(p);
}
__device__ inline v16h pack16(v8h lo, v8h hi) {
    v16h r;
#pragma unroll
    for (int i = 0; i < 8; ++i) { r[i] = lo[i]; r[i + 8] = hi[i]; }
    return r;
}

// ---- CDNA5 async global->LDS copy (16B per lane), tracked by ASYNCcnt ----
__device__ inline void cp_async16(_Float16* lds_dst, const _Float16* gsrc) {
    uint32_t loff = (uint32_t)(uintptr_t)lds_dst;   // low 32 bits = LDS byte addr
    asm volatile("global_load_async_to_lds_b128 %0, %1, off"
                 :: "v"(loff), "v"(gsrc) : "memory");
}
__device__ inline void async_wait_and_barrier() {
    asm volatile("s_wait_asynccnt 0x0" ::: "memory");
    __syncthreads();
}

// One 16x16 output tile per wave; A row-major (M x K), Bt N-major (N x K).
// CDNA5 16-bit A/B fragment layout:
//   lanes 0-15 : row/col = lane,    K chunks {0..7, 16..23}
//   lanes16-31 : row/col = lane-16, K chunks {8..15, 24..31}
template<int K>
__device__ inline v8f wave_gemm_tile(const _Float16* __restrict__ Arow,
                                     const _Float16* __restrict__ Brow,
                                     v8f acc, int lane) {
    const int mn = lane & 15;
    const int kb = (lane >> 4) * 8;
    const _Float16* ap = Arow + (size_t)mn * K + kb;
    const _Float16* bp = Brow + (size_t)mn * K + kb;
#pragma unroll
    for (int k = 0; k < K; k += 32) {
        v16h a = pack16(ld8(ap + k), ld8(ap + k + 16));
        v16h b = pack16(ld8(bp + k), ld8(bp + k + 16));
        acc = __builtin_amdgcn_wmma_f32_16x16x32_f16(
            false, a, false, b, (short)0, acc, false, false);
    }
    return acc;
}

__device__ inline float silu_f(float x) { return x / (1.0f + __expf(-x)); }

// ---------------- layout transform: x(B,C,L) -> seq(NT,C) ----------------
__global__ void k_transpose_in(const float* __restrict__ x, float* __restrict__ seq) {
    int idx = blockIdx.x * blockDim.x + threadIdx.x;
    if (idx >= Bn * Cc * Ll) return;
    int b = idx / (Cc * Ll);
    int r = idx - b * Cc * Ll;
    int c = r / Ll;
    int l = r - c * Ll;
    seq[(size_t)(b * Ll + l) * Cc + c] = x[idx];
}

// ---------------- LayerNorm over C, output f16 ----------------
__global__ void k_layernorm(const float* __restrict__ seq,
                            const float* __restrict__ g, const float* __restrict__ bta,
                            _Float16* __restrict__ out) {
    int t = blockIdx.x * blockDim.x + threadIdx.x;
    if (t >= NT) return;
    const float* row = seq + (size_t)t * Cc;
    float mu = 0.f;
#pragma unroll 4
    for (int c = 0; c < Cc; ++c) mu += row[c];
    mu *= (1.0f / Cc);
    float var = 0.f;
#pragma unroll 4
    for (int c = 0; c < Cc; ++c) { float d = row[c] - mu; var += d * d; }
    var *= (1.0f / Cc);
    float rs = rsqrtf(var + EPSv);
    _Float16* orow = out + (size_t)t * Cc;
#pragma unroll 4
    for (int c = 0; c < Cc; ++c)
        orow[c] = (_Float16)((row[c] - mu) * rs * g[c] + bta[c]);
}

// ---------------- weight converts (to N-major, K-contiguous f16) ----------------
__global__ void k_cvt_inw(const float* __restrict__ w, _Float16* __restrict__ wt) {
    int id = blockIdx.x * blockDim.x + threadIdx.x;
    if (id >= D2IN * Cc) return;
    int n = id / Cc, c = id - n * Cc;
    wt[n * Cc + c] = (_Float16)w[c * D2IN + n];
}
__global__ void k_cvt_outw(const float* __restrict__ w, _Float16* __restrict__ wt) {
    int id = blockIdx.x * blockDim.x + threadIdx.x;
    if (id >= Cc * DIN) return;
    int n = id / DIN, k = id - n * DIN;
    wt[n * DIN + k] = (_Float16)w[k * Cc + n];
}
__global__ void k_cvt_fusw(const float* __restrict__ w, _Float16* __restrict__ wt) {
    int id = blockIdx.x * blockDim.x + threadIdx.x;
    if (id >= Cc * Cc) return;
    int o = id / Cc, c = id - o * Cc;
    wt[o * Cc + c] = (_Float16)w[o * (2 * Cc) + c];
}
__global__ void k_cvt_seq16(const float* __restrict__ s, _Float16* __restrict__ o) {
    int id = blockIdx.x * blockDim.x + threadIdx.x;
    if (id >= NT * Cc) return;
    o[id] = (_Float16)s[id];
}

// ---------------- GEMM 1: xz = h @ in_w  (NT x 96)x(96 x 384) ----------------
// Block = 12 waves (384 thr): one M-tile x half of N (12 N-tiles).
// Weight panel (192 rows x 96 halfs = 36KB) staged via async LDS copy.
__global__ void __launch_bounds__(384)
k_gemm_xz(const _Float16* __restrict__ A, const _Float16* __restrict__ Bt,
          float* __restrict__ D) {
    __shared__ __align__(16) _Float16 Bs[192 * 96];
    int mtile = blockIdx.x >> 1;
    int half  = blockIdx.x & 1;
    int tid   = threadIdx.x;
    const _Float16* src = Bt + (size_t)half * 192 * Cc;
#pragma unroll
    for (int j = 0; j < 6; ++j) {                 // 2304 x 16B total / 384 thr
        int i = tid + j * 384;
        cp_async16(&Bs[i * 8], &src[i * 8]);
    }
    async_wait_and_barrier();

    int wid  = tid >> 5;                          // 0..11
    int lane = tid & 31;
    int ntile = half * 12 + wid;
    v8f acc = {};
    acc = wave_gemm_tile<Cc>(A + (size_t)mtile * 16 * Cc,
                             Bs + (size_t)wid * 16 * Cc, acc, lane);
    int col = ntile * 16 + (lane & 15);
    int rb  = mtile * 16 + 8 * (lane >> 4);
#pragma unroll
    for (int r = 0; r < 8; ++r)
        D[(size_t)(rb + r) * D2IN + col] = acc[r];
}

// ---------------- causal depthwise conv (DCONV=4) + SiLU ----------------
__global__ void k_conv_silu(const float* __restrict__ xz, const float* __restrict__ cw,
                            const float* __restrict__ cb, float* __restrict__ xc) {
    int id = blockIdx.x * blockDim.x + threadIdx.x;
    if (id >= NT * DIN) return;
    int t = id / DIN, d = id - t * DIN;
    int b = t / Ll, l = t - b * Ll;
    float acc = cb[d];
#pragma unroll
    for (int k = 0; k < DCONV; ++k) {
        int lp = l - (DCONV - 1) + k;
        if (lp >= 0)
            acc += cw[d * DCONV + k] * xz[(size_t)(b * Ll + lp) * D2IN + d];
    }
    xc[id] = silu_f(acc);
}

// ---------------- x_proj: dbl = xc @ xproj_w  (K=192, N=38) ----------------
__global__ void k_xproj(const float* __restrict__ xc, const float* __restrict__ w,
                        float* __restrict__ dbl) {
    int id = blockIdx.x * blockDim.x + threadIdx.x;
    if (id >= NT * XPJ) return;
    int t = id / XPJ, j = id - t * XPJ;
    const float* row = xc + (size_t)t * DIN;
    float acc = 0.f;
#pragma unroll 4
    for (int d = 0; d < DIN; ++d) acc += row[d] * w[d * XPJ + j];
    dbl[id] = acc;
}

// ---------------- delta = softplus(dt @ dt_w + dt_b) (K=6) ----------------
__global__ void k_delta(const float* __restrict__ dbl, const float* __restrict__ w,
                        const float* __restrict__ bb, float* __restrict__ delta) {
    int id = blockIdx.x * blockDim.x + threadIdx.x;
    if (id >= NT * DIN) return;
    int t = id / DIN, d = id - t * DIN;
    const float* row = dbl + (size_t)t * XPJ;
    float acc = bb[d];
#pragma unroll
    for (int r = 0; r < DTRANK; ++r) acc += row[r] * w[r * DIN + d];
    delta[id] = (acc > 20.f) ? acc : log1pf(__expf(acc));
}

// ======== chunked selective scan (associative, 3 passes, 32x parallelism) =====
// Wave layout (all passes): 2 channels x 16 states; lanes s=0..15 per half-wave.
// wid = ((b*96 + pair)*NCH + ch)

// Pass 1: per-chunk (prod dA, local h end), h_init = 0
__global__ void k_scan1(const float* __restrict__ delta, const float* __restrict__ dbl,
                        const float* __restrict__ xc, const float* __restrict__ A_log,
                        float* __restrict__ cA, float* __restrict__ cH) {
    int wid = blockIdx.x * (blockDim.x >> 5) + (threadIdx.x >> 5);   // 0..12287
    int lane = threadIdx.x & 31;
    int ch = wid & (NCH - 1);
    int rest = wid >> 5;                 // NCH==32
    int b = rest / (DIN / 2);
    int pair = rest - b * (DIN / 2);
    int sub = lane >> 4;
    int s = lane & 15;
    int d = pair * 2 + sub;
    float negA = -__expf(A_log[d * DSTATE + s]);
    float aprod = 1.f, hst = 0.f;
    int l0 = ch * CH;
    for (int l = l0; l < l0 + CH; ++l) {
        int t = b * Ll + l;
        float dl = delta[(size_t)t * DIN + d];
        float xv = xc[(size_t)t * DIN + d];
        float Bv = dbl[(size_t)t * XPJ + DTRANK + s];
        float dA = __expf(dl * negA);
        hst = fmaf(dA, hst, dl * Bv * xv);
        aprod *= dA;
    }
    size_t idx = ((size_t)(b * DIN + d) * DSTATE + s) * NCH + ch;
    cA[idx] = aprod;
    cH[idx] = hst;
}

// Pass 2: sequential combine over 32 chunks per (b,d,s) -> carry-in per chunk
__global__ void k_carry(const float* __restrict__ cA, const float* __restrict__ cH,
                        float* __restrict__ cIn) {
    int id = blockIdx.x * blockDim.x + threadIdx.x;    // Bn*DIN*DSTATE = 12288
    if (id >= Bn * DIN * DSTATE) return;
    size_t base = (size_t)id * NCH;
    float carry = 0.f;
#pragma unroll
    for (int ch = 0; ch < NCH; ++ch) {
        cIn[base + ch] = carry;
        carry = cH[base + ch] + cA[base + ch] * carry;
    }
}

// Pass 3: re-run chunk from carry-in; cross-state reduce; SiLU(z) gate; y16 out
__global__ void k_scan2(const float* __restrict__ delta, const float* __restrict__ dbl,
                        const float* __restrict__ xc, const float* __restrict__ xz,
                        const float* __restrict__ A_log, const float* __restrict__ Dp,
                        const float* __restrict__ cIn, _Float16* __restrict__ y16) {
    int wid = blockIdx.x * (blockDim.x >> 5) + (threadIdx.x >> 5);
    int lane = threadIdx.x & 31;
    int ch = wid & (NCH - 1);
    int rest = wid >> 5;
    int b = rest / (DIN / 2);
    int pair = rest - b * (DIN / 2);
    int sub = lane >> 4;
    int s = lane & 15;
    int d = pair * 2 + sub;
    float negA = -__expf(A_log[d * DSTATE + s]);
    float Dpd = Dp[d];
    float hst = cIn[((size_t)(b * DIN + d) * DSTATE + s) * NCH + ch];
    int l0 = ch * CH;
    for (int l = l0; l < l0 + CH; ++l) {
        int t = b * Ll + l;
        float dl = delta[(size_t)t * DIN + d];
        float xv = xc[(size_t)t * DIN + d];
        float Bv = dbl[(size_t)t * XPJ + DTRANK + s];
        float Cv = dbl[(size_t)t * XPJ + DTRANK + DSTATE + s];
        float dA = __expf(dl * negA);
        hst = fmaf(dA, hst, dl * Bv * xv);
        float p = hst * Cv;
        p += __shfl_xor(p, 1);
        p += __shfl_xor(p, 2);
        p += __shfl_xor(p, 4);
        p += __shfl_xor(p, 8);
        if (s == 0) {
            float yv = p + Dpd * xv;
            float zv = xz[(size_t)t * D2IN + DIN + d];
            y16[(size_t)t * DIN + d] = (_Float16)(yv * silu_f(zv));
        }
    }
}

// ---------------- GEMM 2: seq += y @ out_w (residual fused into C) ----------
// Block = 6 waves: one M-tile x all 6 N-tiles; full weight (36KB) async-staged.
__global__ void __launch_bounds__(192)
k_gemm_out_res(const _Float16* __restrict__ A, const _Float16* __restrict__ Bt,
               float* __restrict__ seq) {
    __shared__ __align__(16) _Float16 Bs[Cc * DIN];
    int mtile = blockIdx.x;
    int tid = threadIdx.x;
#pragma unroll
    for (int j = 0; j < 12; ++j) {                // 2304 x 16B / 192 thr
        int i = tid + j * 192;
        cp_async16(&Bs[i * 8], &Bt[i * 8]);
    }
    async_wait_and_barrier();

    int ntile = tid >> 5;
    int lane = tid & 31;
    int col = ntile * 16 + (lane & 15);
    int rb  = mtile * 16 + 8 * (lane >> 4);
    v8f acc;
#pragma unroll
    for (int r = 0; r < 8; ++r) acc[r] = seq[(size_t)(rb + r) * Cc + col];
    acc = wave_gemm_tile<DIN>(A + (size_t)mtile * 16 * DIN,
                              Bs + (size_t)ntile * 16 * DIN, acc, lane);
#pragma unroll
    for (int r = 0; r < 8; ++r)
        seq[(size_t)(rb + r) * Cc + col] = acc[r];
}

// ---------------- global-context small kernels ----------------
__global__ void k_xmean(const float* __restrict__ x, float* __restrict__ xm) {
    __shared__ float sm[256];
    int bc = blockIdx.x;
    const float* p = x + (size_t)bc * Ll;
    float acc = 0.f;
    for (int l = threadIdx.x; l < Ll; l += blockDim.x) acc += p[l];
    sm[threadIdx.x] = acc;
    __syncthreads();
    for (int off = 128; off > 0; off >>= 1) {
        if (threadIdx.x < off) sm[threadIdx.x] += sm[threadIdx.x + off];
        __syncthreads();
    }
    if (threadIdx.x == 0) xm[bc] = sm[0] * (1.0f / Ll);
}
__global__ void k_gp(const float* __restrict__ xm, const float* __restrict__ w,
                     const float* __restrict__ bb, float* __restrict__ gp) {
    int id = blockIdx.x * blockDim.x + threadIdx.x;
    if (id >= Bn * Cc) return;
    int b = id / Cc, o = id - b * Cc;
    float acc = bb[o];
#pragma unroll 4
    for (int c = 0; c < Cc; ++c) acc += xm[b * Cc + c] * w[o * Cc + c];
    gp[id] = fmaxf(acc, 0.f);
}
__global__ void k_bias2(const float* __restrict__ gp, const float* __restrict__ fw,
                        const float* __restrict__ fb, float* __restrict__ b2) {
    int id = blockIdx.x * blockDim.x + threadIdx.x;
    if (id >= Bn * Cc) return;
    int b = id / Cc, o = id - b * Cc;
    float acc = fb[o];
#pragma unroll 4
    for (int c = 0; c < Cc; ++c) acc += fw[o * (2 * Cc) + Cc + c] * gp[b * Cc + c];
    b2[id] = acc;
}

// ---------------- GEMM 3: out_pre = seq16 @ fus_w[:, :96]^T + bias2[b,o] ------
__global__ void __launch_bounds__(192)
k_gemm_fuse(const _Float16* __restrict__ A, const _Float16* __restrict__ Bt,
            const float* __restrict__ b2, float* __restrict__ D) {
    __shared__ __align__(16) _Float16 Bs[Cc * Cc];
    int mtile = blockIdx.x;
    int tid = threadIdx.x;
#pragma unroll
    for (int j = 0; j < 6; ++j) {                 // 1152 x 16B / 192 thr
        int i = tid + j * 192;
        cp_async16(&Bs[i * 8], &Bt[i * 8]);
    }
    async_wait_and_barrier();

    int ntile = tid >> 5;
    int lane = tid & 31;
    int col = ntile * 16 + (lane & 15);
    int rb  = mtile * 16 + 8 * (lane >> 4);
    int b   = (mtile * 16) >> 12;                 // tiles never cross batch
    v8f acc;
    float bias = b2[b * Cc + col];
#pragma unroll
    for (int r = 0; r < 8; ++r) acc[r] = bias;
    acc = wave_gemm_tile<Cc>(A + (size_t)mtile * 16 * Cc,
                             Bs + (size_t)ntile * 16 * Cc, acc, lane);
#pragma unroll
    for (int r = 0; r < 8; ++r)
        D[(size_t)(rb + r) * Cc + col] = acc[r];
}

// ---------------- BatchNorm stats over (B,H,W) per channel ----------------
__global__ void k_bnstats(const float* __restrict__ op, float* __restrict__ mu,
                          float* __restrict__ vr) {
    __shared__ float s1[256], s2[256];
    int c = blockIdx.x;
    float a = 0.f, q = 0.f;
    for (int t = threadIdx.x; t < NT; t += blockDim.x) {
        float v = op[(size_t)t * Cc + c];
        a += v; q += v * v;
    }
    s1[threadIdx.x] = a; s2[threadIdx.x] = q;
    __syncthreads();
    for (int off = 128; off > 0; off >>= 1) {
        if (threadIdx.x < off) {
            s1[threadIdx.x] += s1[threadIdx.x + off];
            s2[threadIdx.x] += s2[threadIdx.x + off];
        }
        __syncthreads();
    }
    if (threadIdx.x == 0) {
        float m = s1[0] * (1.0f / NT);
        mu[c] = m;
        vr[c] = s2[0] * (1.0f / NT) - m * m;
    }
}

// ---------------- final: BN affine + residual, write (B,C,H,W) ----------------
__global__ void k_final(const float* __restrict__ op, const float* __restrict__ mu,
                        const float* __restrict__ vr, const float* __restrict__ g,
                        const float* __restrict__ bb, const float* __restrict__ x,
                        float* __restrict__ out) {
    int idx = blockIdx.x * blockDim.x + threadIdx.x;
    if (idx >= Bn * Cc * Ll) return;
    int b = idx / (Cc * Ll);
    int r = idx - b * Cc * Ll;
    int c = r / Ll;
    int l = r - c * Ll;
    int t = b * Ll + l;
    float v = (op[(size_t)t * Cc + c] - mu[c]) * rsqrtf(vr[c] + EPSv) * g[c] + bb[c];
    out[idx] = v + x[idx];
}

// =======================================================================
static inline size_t alignup(size_t v) { return (v + 255) & ~(size_t)255; }

extern "C" void kernel_launch(void* const* d_in, const int* in_sizes, int n_in,
                              void* d_out, int out_size, void* d_ws, size_t ws_size,
                              hipStream_t stream) {
    const float* x       = (const float*)d_in[0];
    const float* ln_g    = (const float*)d_in[1];
    const float* ln_b    = (const float*)d_in[2];
    const float* in_w    = (const float*)d_in[3];
    const float* conv_w  = (const float*)d_in[4];
    const float* conv_b  = (const float*)d_in[5];
    const float* xproj_w = (const float*)d_in[6];
    const float* dt_w    = (const float*)d_in[7];
    const float* dt_b    = (const float*)d_in[8];
    const float* A_log   = (const float*)d_in[9];
    const float* Dp      = (const float*)d_in[10];
    const float* out_w   = (const float*)d_in[11];
    const float* gp_w    = (const float*)d_in[12];
    const float* gp_b    = (const float*)d_in[13];
    const float* fus_w   = (const float*)d_in[14];
    const float* fus_b   = (const float*)d_in[15];
    const float* bn_g    = (const float*)d_in[16];
    const float* bn_b    = (const float*)d_in[17];
    float* out = (float*)d_out;

    // -------- workspace carve-out --------
    char* p = (char*)d_ws;
    float*    seq    = (float*)p;       p += alignup((size_t)NT * Cc * 4);
    _Float16* h16    = (_Float16*)p;    p += alignup((size_t)NT * Cc * 2);
    float*    xz     = (float*)p;       p += alignup((size_t)NT * D2IN * 4);
    float*    xc     = (float*)p;       p += alignup((size_t)NT * DIN * 4);
    float*    dbl    = (float*)p;       p += alignup((size_t)NT * XPJ * 4);
    float*    delta  = (float*)p;       p += alignup((size_t)NT * DIN * 4);
    _Float16* y16    = (_Float16*)p;    p += alignup((size_t)NT * DIN * 2);
    float*    opre   = (float*)p;       p += alignup((size_t)NT * Cc * 4);
    _Float16* inw16  = (_Float16*)p;    p += alignup((size_t)D2IN * Cc * 2);
    _Float16* outw16 = (_Float16*)p;    p += alignup((size_t)Cc * DIN * 2);
    _Float16* fus16  = (_Float16*)p;    p += alignup((size_t)Cc * Cc * 2);
    float*    xm     = (float*)p;       p += alignup((size_t)Bn * Cc * 4);
    float*    gp     = (float*)p;       p += alignup((size_t)Bn * Cc * 4);
    float*    b2     = (float*)p;       p += alignup((size_t)Bn * Cc * 4);
    float*    mu     = (float*)p;       p += alignup(96 * 4);
    float*    vr     = (float*)p;       p += alignup(96 * 4);
    float*    cA     = (float*)p;       p += alignup((size_t)Bn * DIN * DSTATE * NCH * 4);
    float*    cHh    = (float*)p;       p += alignup((size_t)Bn * DIN * DSTATE * NCH * 4);
    float*    cIn    = (float*)p;       p += alignup((size_t)Bn * DIN * DSTATE * NCH * 4);
    (void)ws_size; (void)n_in; (void)in_sizes; (void)out_size;

    const int T = 256;
    auto grid = [](long n, int t) { return (unsigned)((n + t - 1) / t); };

    k_transpose_in<<<grid((long)Bn * Cc * Ll, T), T, 0, stream>>>(x, seq);

    for (int i = 0; i < NLAYERS; ++i) {
        k_layernorm<<<grid(NT, T), T, 0, stream>>>(seq, ln_g + i * Cc, ln_b + i * Cc, h16);
        k_cvt_inw<<<grid((long)D2IN * Cc, T), T, 0, stream>>>(in_w + (size_t)i * Cc * D2IN, inw16);
        // 1024 M-tiles x 2 N-halves, 12 waves/block
        k_gemm_xz<<<2048, 384, 0, stream>>>(h16, inw16, xz);
        k_conv_silu<<<grid((long)NT * DIN, T), T, 0, stream>>>(
            xz, conv_w + (size_t)i * DIN * DCONV, conv_b + (size_t)i * DIN, xc);
        k_xproj<<<grid((long)NT * XPJ, T), T, 0, stream>>>(
            xc, xproj_w + (size_t)i * DIN * XPJ, dbl);
        k_delta<<<grid((long)NT * DIN, T), T, 0, stream>>>(
            dbl, dt_w + (size_t)i * DTRANK * DIN, dt_b + (size_t)i * DIN, delta);
        // chunked scan: 12288 waves -> 1536 blocks of 8 waves
        k_scan1<<<1536, 256, 0, stream>>>(delta, dbl, xc,
                                          A_log + (size_t)i * DIN * DSTATE, cA, cHh);
        k_carry<<<48, 256, 0, stream>>>(cA, cHh, cIn);
        k_scan2<<<1536, 256, 0, stream>>>(delta, dbl, xc, xz,
                                          A_log + (size_t)i * DIN * DSTATE,
                                          Dp + (size_t)i * DIN, cIn, y16);
        k_cvt_outw<<<grid((long)Cc * DIN, T), T, 0, stream>>>(
            out_w + (size_t)i * DIN * Cc, outw16);
        // 1024 M-tiles, 6 waves/block
        k_gemm_out_res<<<1024, 192, 0, stream>>>(y16, outw16, seq);
    }

    // global context branch
    k_xmean<<<Bn * Cc, 256, 0, stream>>>(x, xm);
    k_gp<<<grid(Bn * Cc, T), T, 0, stream>>>(xm, gp_w, gp_b, gp);
    k_bias2<<<grid(Bn * Cc, T), T, 0, stream>>>(gp, fus_w, fus_b, b2);

    // fusion GEMM
    k_cvt_seq16<<<grid((long)NT * Cc, T), T, 0, stream>>>(seq, h16);
    k_cvt_fusw<<<grid((long)Cc * Cc, T), T, 0, stream>>>(fus_w, fus16);
    k_gemm_fuse<<<1024, 192, 0, stream>>>(h16, fus16, b2, opre);

    // batchnorm + residual
    k_bnstats<<<Cc, 256, 0, stream>>>(opre, mu, vr);
    k_final<<<grid((long)Bn * Cc * Ll, T), T, 0, stream>>>(opre, mu, vr, bn_g, bn_b, x, out);
}
